// GCN_38938173505764
// MI455X (gfx1250) — compile-verified, matrix operand order checked by reading, not run on hardware
//
#include <hip/hip_runtime.h>
#include <hip/hip_fp16.h>

typedef __attribute__((ext_vector_type(16))) _Float16 v16h;
typedef __attribute__((ext_vector_type(8)))  float    v8f;

#define N_NODES 100000
#define N_EDGES 1600000
#define NFEAT   128
#define NHID    64
#define NCLASS  16

// ---------------------------------------------------------------------------
// Degree / normalization: deg starts at 1 (self loop), +1 per incoming edge,
// then dinv = 1/sqrt(deg) in place.
// ---------------------------------------------------------------------------
__global__ void k_init_deg(float* __restrict__ deg) {
    int i = blockIdx.x * blockDim.x + threadIdx.x;
    if (i < N_NODES) deg[i] = 1.0f;
}

__global__ void k_edge_deg(const int* __restrict__ col, float* __restrict__ deg) {
    int e = blockIdx.x * blockDim.x + threadIdx.x;
    if (e < N_EDGES) atomicAdd(&deg[col[e]], 1.0f);
}

__global__ void k_deg_to_dinv(float* __restrict__ deg) {
    int i = blockIdx.x * blockDim.x + threadIdx.x;
    if (i < N_NODES) {
        float d = deg[i];
        deg[i] = d > 0.0f ? rsqrtf(d) : 0.0f;
    }
}

// ---------------------------------------------------------------------------
// Pre-pack W[K x N] (f32, row-major) into f16 WMMA B-fragments, one 32-byte
// contiguous chunk per (n-tile, k-chunk, lane). Fragment layout per CDNA5 ISA
// 7.12.2: N = lane&15; K striping {kb..kb+7, 16+kb..16+kb+7}, kb = 8*(lane>=16).
// Linear index: ((tn*(K/32) + kc)*32 + lane)*16 + i.
// ---------------------------------------------------------------------------
template <int K, int N>
__global__ void k_pack_w(const float* __restrict__ W, _Float16* __restrict__ P) {
    int t = blockIdx.x * blockDim.x + threadIdx.x;
    if (t >= K * N) return;
    int i    = t & 15;
    int u    = t >> 4;
    int lane = u & 31;
    int v    = u >> 5;
    int kc   = v % (K / 32);
    int tn   = v / (K / 32);
    int l16  = lane & 15;
    int kb   = (lane >> 4) * 8;
    int k    = (i < 8) ? (kb + i) : (16 + kb + (i - 8));
    P[t] = (_Float16)W[(long)(kc * 32 + k) * N + tn * 16 + l16];
}

// ---------------------------------------------------------------------------
// WMMA GEMM: C[M x N] = op(A[M x K]) * Bpacked, f16 matrix core, f32 accum.
// One wave owns a full 16 x N output strip: the A fragment (the only
// HBM-resident operand) is loaded and converted ONCE per k-chunk and reused
// across all N/16 accumulators; B fragments are single contiguous v16h loads
// from the pre-packed (L1-resident) weight buffer. RELU fuses relu(h) @ W2.
// ---------------------------------------------------------------------------
template <int K, int N, bool RELU>
__global__ void k_gemm_wmma(const float* __restrict__ A,
                            const _Float16* __restrict__ Bp,
                            float* __restrict__ C, int m_tiles) {
    constexpr int NT = N / 16;   // n-tiles per wave
    constexpr int KC = K / 32;   // k-chunks
    const int wave = (blockIdx.x * blockDim.x + threadIdx.x) >> 5;
    const int lane = threadIdx.x & 31;
    if (wave >= m_tiles) return;

    const int l16 = lane & 15;
    const int kb  = (lane >> 4) * 8;                       // per-lane K sub-offset
    const float* __restrict__ arow = A + (long)(wave * 16 + l16) * K;

    v8f acc[NT] = {};
#pragma unroll
    for (int kc = 0; kc < KC; ++kc) {
        v16h a;
#pragma unroll
        for (int i = 0; i < 8; ++i) {
            float a0 = arow[kc * 32 + kb + i];
            float a1 = arow[kc * 32 + 16 + kb + i];
            if (RELU) { a0 = fmaxf(a0, 0.0f); a1 = fmaxf(a1, 0.0f); }
            a[i]     = (_Float16)a0;
            a[8 + i] = (_Float16)a1;
        }
#pragma unroll
        for (int tn = 0; tn < NT; ++tn) {
            v16h b = *(const v16h*)(Bp + ((long)(tn * KC + kc) * 32 + lane) * 16);
            // v_wmma_f32_16x16x32_f16: (neg_a, A, neg_b, B, c_mod, C, reuse_a, reuse_b)
            acc[tn] = __builtin_amdgcn_wmma_f32_16x16x32_f16(false, a, false, b,
                                                             (short)0, acc[tn],
                                                             false, false);
        }
    }

    // C/D layout: VGPR r -> row r + 8*(lane>=16), col = lane&15
    float* __restrict__ crow = C + ((long)wave * 16 + (lane >> 4) * 8) * N + l16;
#pragma unroll
    for (int tn = 0; tn < NT; ++tn)
#pragma unroll
        for (int r = 0; r < 8; ++r) crow[tn * 16 + (long)r * N] = acc[tn][r];
}

// ---------------------------------------------------------------------------
// h = bias + dinv[i]^2 * xw   (bias + self-loop term, no atomics needed)
// ---------------------------------------------------------------------------
template <int F>
__global__ void k_init_agg(const float* __restrict__ xw,
                           const float* __restrict__ dinv,
                           const float* __restrict__ bias,
                           float* __restrict__ h) {
    long t = (long)blockIdx.x * blockDim.x + threadIdx.x;
    if (t >= (long)N_NODES * F) return;
    int i = (int)(t / F);
    int j = (int)(t - (long)i * F);
    float di = dinv[i];
    h[t] = bias[j] + di * di * xw[t];
}

// ---------------------------------------------------------------------------
// Layer-1 edge scatter: one wave per edge, 64 features -> 2 floats per lane.
// xw and h (25.6 MB each) are L2-resident so gathers + f32 atomics stay
// on-chip.
// ---------------------------------------------------------------------------
__global__ void k_edge_scatter64(const int* __restrict__ row,
                                 const int* __restrict__ col,
                                 const float* __restrict__ dinv,
                                 const float* __restrict__ xw,
                                 float* __restrict__ h) {
    const int e    = (blockIdx.x * blockDim.x + threadIdx.x) >> 5;
    const int lane = threadIdx.x & 31;
    if (e >= N_EDGES) return;
    const int r = row[e];
    const int c = col[e];
    const float w = dinv[r] * dinv[c];
    const float* __restrict__ src = xw + (long)r * NHID;
    float* __restrict__ dst       = h  + (long)c * NHID;
    atomicAdd(&dst[lane],      w * src[lane]);
    atomicAdd(&dst[lane + 32], w * src[lane + 32]);
}

// ---------------------------------------------------------------------------
// Layer-2 edge scatter: 16 classes -> half-wave per edge (2 edges per wave).
// ---------------------------------------------------------------------------
__global__ void k_edge_scatter16(const int* __restrict__ row,
                                 const int* __restrict__ col,
                                 const float* __restrict__ dinv,
                                 const float* __restrict__ h2,
                                 float* __restrict__ out) {
    const int wave = (blockIdx.x * blockDim.x + threadIdx.x) >> 5;
    const int lane = threadIdx.x & 31;
    const int e    = wave * 2 + (lane >> 4);
    const int j    = lane & 15;
    if (e >= N_EDGES) return;
    const int r = row[e];
    const int c = col[e];
    const float w = dinv[r] * dinv[c];
    atomicAdd(&out[(long)c * NCLASS + j], w * h2[(long)r * NCLASS + j]);
}

// ---------------------------------------------------------------------------
// In-place log_softmax over 16 classes per node.
// ---------------------------------------------------------------------------
__global__ void k_log_softmax16(float* __restrict__ out) {
    int i = blockIdx.x * blockDim.x + threadIdx.x;
    if (i >= N_NODES) return;
    float* __restrict__ rowp = out + (long)i * NCLASS;
    float v[NCLASS];
    float m = -3.402823466e38f;
#pragma unroll
    for (int j = 0; j < NCLASS; ++j) { v[j] = rowp[j]; m = fmaxf(m, v[j]); }
    float s = 0.0f;
#pragma unroll
    for (int j = 0; j < NCLASS; ++j) s += __expf(v[j] - m);
    float lse = m + __logf(s);
#pragma unroll
    for (int j = 0; j < NCLASS; ++j) rowp[j] = v[j] - lse;
}

// ---------------------------------------------------------------------------
// Launch
// ---------------------------------------------------------------------------
extern "C" void kernel_launch(void* const* d_in, const int* in_sizes, int n_in,
                              void* d_out, int out_size, void* d_ws, size_t ws_size,
                              hipStream_t stream) {
    const float* x   = (const float*)d_in[0];
    const int*   ei  = (const int*)d_in[1];      // [2, E]: row = src, col = dst
    const float* W1  = (const float*)d_in[2];
    const float* b1  = (const float*)d_in[3];
    const float* W2  = (const float*)d_in[4];
    const float* b2  = (const float*)d_in[5];
    float* out       = (float*)d_out;            // [N_NODES, NCLASS]

    const int* row = ei;
    const int* col = ei + N_EDGES;

    // Workspace layout (256B aligned):
    //   dinv : N_NODES floats                       (0.4 MB)
    //   W1p  : NFEAT*NHID f16, B-fragment order     (16 KB)
    //   W2p  : NHID*NCLASS f16, B-fragment order    (2 KB)
    //   bufA : xw [N,64] then reused as h2 [N,16]   (25.6 MB)
    //   h    : [N,64]                               (25.6 MB)
    auto align256 = [](size_t v) { return (v + 255) & ~(size_t)255; };
    char* ws = (char*)d_ws;
    size_t o = 0;
    float*     dinv = (float*)(ws + o);     o += align256((size_t)N_NODES * 4);
    _Float16*  W1p  = (_Float16*)(ws + o);  o += align256((size_t)NFEAT * NHID * 2);
    _Float16*  W2p  = (_Float16*)(ws + o);  o += align256((size_t)NHID * NCLASS * 2);
    float*     bufA = (float*)(ws + o);     o += align256((size_t)N_NODES * NHID * 4);
    float*     h    = (float*)(ws + o);
    float*     h2   = bufA;                 // reuse xw space after layer-1 scatter

    const int B = 256;

    // 0) pack weights into WMMA B-fragment layout (tiny, one-shot)
    k_pack_w<NFEAT, NHID><<<(NFEAT * NHID + B - 1) / B, B, 0, stream>>>(W1, W1p);
    k_pack_w<NHID, NCLASS><<<(NHID * NCLASS + B - 1) / B, B, 0, stream>>>(W2, W2p);

    // 1) symmetric normalization dinv = deg^{-1/2}
    k_init_deg<<<(N_NODES + B - 1) / B, B, 0, stream>>>(dinv);
    k_edge_deg<<<(N_EDGES + B - 1) / B, B, 0, stream>>>(col, dinv);
    k_deg_to_dinv<<<(N_NODES + B - 1) / B, B, 0, stream>>>(dinv);

    const int m_tiles = N_NODES / 16;        // 6250, exact

    // 2) xw = x @ W1  (one wave per 16x64 strip)
    k_gemm_wmma<NFEAT, NHID, false><<<(m_tiles * 32 + B - 1) / B, B, 0, stream>>>(
        x, W1p, bufA, m_tiles);

    // 3) h = b1 + dinv^2 * xw, then scatter edges
    k_init_agg<NHID><<<((long)N_NODES * NHID + B - 1) / B, B, 0, stream>>>(bufA, dinv, b1, h);
    k_edge_scatter64<<<((long)N_EDGES * 32 + B - 1) / B, B, 0, stream>>>(row, col, dinv, bufA, h);

    // 4) h2 = relu(h) @ W2  (relu fused into A-operand load)
    k_gemm_wmma<NHID, NCLASS, true><<<(m_tiles * 32 + B - 1) / B, B, 0, stream>>>(
        h, W2p, h2, m_tiles);

    // 5) out = b2 + dinv^2 * h2, scatter edges, log_softmax in place
    k_init_agg<NCLASS><<<((long)N_NODES * NCLASS + B - 1) / B, B, 0, stream>>>(h2, dinv, b2, out);
    k_edge_scatter16<<<((long)N_EDGES / 2 * 32 + B - 1) / B, B, 0, stream>>>(row, col, dinv, h2, out);
    k_log_softmax16<<<(N_NODES + B - 1) / B, B, 0, stream>>>(out);
}
